// RecurrentRGCN_39513699123403
// MI455X (gfx1250) — matrix operand order repaired and checked by reading, not compiled, not run on hardware
//
#include <hip/hip_runtime.h>
#include <math.h>

typedef __attribute__((ext_vector_type(16))) _Float16 v16h;
typedef __attribute__((ext_vector_type(8)))  _Float16 v8h;
typedef __attribute__((ext_vector_type(8)))  float    v8f;

#define N_ENT  20000
#define HDIM   256
#define TSEQ   8
#define NEDGE  100000
#define NRTE   150000
#define NREL   500
#define NRELP  512
#define RRELU_SLOPE 0.22916666666666666f

__device__ __forceinline__ float sigf(float x) { return 1.0f / (1.0f + __expf(-x)); }

__device__ __forceinline__ void atomic_addf(float* p, float v) {
    __hip_atomic_fetch_add(p, v, __ATOMIC_RELAXED, __HIP_MEMORY_SCOPE_AGENT);
}

// ---------------------------------------------------------------------------
// WMMA GEMM: C[M,NC] = A[M,K](f16,row-major,LDA) * Bp (packed B fragments).
// Packed layout: elem (k,n) -> t = ((k>>5)*(NC/16) + (n>>4))*512
//                                + (((k>>4)&1)*16 + (n&15))*16 + (k&15)
// Each wave computes 4 adjacent 16x16 n-tiles (4 independent accumulators ->
// no WMMA->WMMA RAW chain; A fragment reused 4x; B loads are 4 contiguous
// 32B chunks). Block = (32,4) waves => 256 output columns per block.
// A must have >= gridDim.y*16 addressable rows (zero-padded where needed).
// ---------------------------------------------------------------------------
template<int NC, int K, int LDA, int LDC>
__global__ void wmma_gemm4_kernel(const _Float16* __restrict__ A,
                                  const _Float16* __restrict__ Bp,
                                  float* __restrict__ C, int M) {
    constexpr int NT = NC >> 4;
    const int lane = threadIdx.x;                       // 0..31
    const int nt0  = (blockIdx.x * 4 + threadIdx.y) * 4; // first of 4 n-tiles
    const int m0   = blockIdx.y << 4;
    const int g    = lane >> 4;
    const int ml   = lane & 15;

    v8f acc0 = {}, acc1 = {}, acc2 = {}, acc3 = {};
    const _Float16* arow  = A  + (size_t)(m0 + ml) * LDA;
    const _Float16* bbase = Bp + ((size_t)nt0 * 32 + lane) * 16;

    for (int k0 = 0; k0 < K; k0 += 32) {
        v8h alo = *(const v8h*)(arow + k0 + g * 16);
        v8h ahi = *(const v8h*)(arow + k0 + g * 16 + 8);
        v16h a;
#pragma unroll
        for (int i = 0; i < 8; ++i) { a[i] = alo[i]; a[8 + i] = ahi[i]; }

        const _Float16* bk = bbase + (size_t)(k0 >> 5) * NT * 512;
        v16h b0 = *(const v16h*)(bk);
        v16h b1 = *(const v16h*)(bk + 512);
        v16h b2 = *(const v16h*)(bk + 1024);
        v16h b3 = *(const v16h*)(bk + 1536);
        acc0 = __builtin_amdgcn_wmma_f32_16x16x32_f16(false, a, false, b0, (short)0, acc0, false, false);
        acc1 = __builtin_amdgcn_wmma_f32_16x16x32_f16(false, a, false, b1, (short)0, acc1, false, false);
        acc2 = __builtin_amdgcn_wmma_f32_16x16x32_f16(false, a, false, b2, (short)0, acc2, false, false);
        acc3 = __builtin_amdgcn_wmma_f32_16x16x32_f16(false, a, false, b3, (short)0, acc3, false, false);
    }

    const int mb = m0 + (g << 3);
#pragma unroll
    for (int j = 0; j < 8; ++j) {
        int m = mb + j;
        if (m < M) {
            float* cr = C + (size_t)m * LDC + ml;
            cr[(nt0 + 0) * 16] = acc0[j];
            cr[(nt0 + 1) * 16] = acc1[j];
            cr[(nt0 + 2) * 16] = acc2[j];
            cr[(nt0 + 3) * 16] = acc3[j];
        }
    }
}

// Pack an f32 weight block into a combined WMMA B-fragment layout (f16).
// transpose=1: B(k, ncolOff+nl) = W[nl*K + k]  (W is [ncols,K], for x @ W.T)
// transpose=0: B(k, ncolOff+nl) = W[k*ncols + nl]
__global__ void pack_b_kernel(const float* __restrict__ W, _Float16* __restrict__ Bp,
                              int K, int NCtot, int ncolOff, int ncols, int transpose) {
    int idx = blockIdx.x * 256 + threadIdx.x;
    if (idx >= K * ncols) return;
    int k  = idx / ncols;
    int nl = idx - k * ncols;
    int n  = ncolOff + nl;
    float v = transpose ? W[(size_t)nl * K + k] : W[(size_t)k * ncols + nl];
    int tile = (k >> 5) * (NCtot >> 4) + (n >> 4);
    int lane = (((k >> 4) & 1) << 4) | (n & 15);
    int i    = k & 15;
    Bp[(size_t)tile * 512 + lane * 16 + i] = (_Float16)v;
}

// ---------------------------------------------------------------------------
__global__ void init_ent_kernel(const float* __restrict__ ent, float* __restrict__ h,
                                _Float16* __restrict__ hf) {
    __shared__ float red[256];
    int n = blockIdx.x, j = threadIdx.x;
    float v = ent[(size_t)n * HDIM + j];
    red[j] = v * v;
    __syncthreads();
    for (int s = 128; s > 0; s >>= 1) { if (j < s) red[j] += red[j + s]; __syncthreads(); }
    float inv = 1.0f / fmaxf(sqrtf(red[0]), 1e-12f);
    float o = v * inv;
    h[(size_t)n * HDIM + j]  = o;
    hf[(size_t)n * HDIM + j] = (_Float16)o;
}

__global__ void init_rel_kernel(const float* __restrict__ emb_rel, float* __restrict__ h0,
                                _Float16* __restrict__ h0f) {
    int r = blockIdx.x, j = threadIdx.x;
    float v = emb_rel[(size_t)r * HDIM + j];
    h0[(size_t)r * HDIM + j]  = v;
    h0f[(size_t)r * HDIM + j] = (_Float16)v;
}

// Grid-stride segment accumulation with next-row prefetch.
__global__ void seg_accum_kernel(const float* __restrict__ h, const int* __restrict__ rte,
                                 const int* __restrict__ rrel, float* __restrict__ sums,
                                 float* __restrict__ cnts) {
    int j = threadIdx.x;
    for (int m = blockIdx.x; m < NRTE; m += gridDim.x) {
        int mn = m + gridDim.x;
        if (mn < NRTE)
            __builtin_prefetch(&h[(size_t)rte[mn] * HDIM + j], 0, 3);
        int e = rte[m], r = rrel[m];
        atomic_addf(&sums[(size_t)r * HDIM + j], h[(size_t)e * HDIM + j]);
        if (j == 0) atomic_addf(&cnts[r], 1.0f);
    }
}

// x_in = [emb_rel | sums/max(cnt,1)] in f16 (rows < NREL; pad rows stay 0)
__global__ void build_xin_kernel(const float* __restrict__ emb_rel,
                                 const float* __restrict__ sums,
                                 const float* __restrict__ cnts,
                                 _Float16* __restrict__ xinf) {
    int r = blockIdx.x, j = threadIdx.x;
    float c = fmaxf(cnts[r], 1.0f);
    xinf[(size_t)r * (2 * HDIM) + j]        = (_Float16)emb_rel[(size_t)r * HDIM + j];
    xinf[(size_t)r * (2 * HDIM) + HDIM + j] = (_Float16)(sums[(size_t)r * HDIM + j] / c);
}

// GRU gates + row l2norm -> h0 (f32) and h0f (f16)
__global__ void gru_pw_kernel(const float* __restrict__ gi, const float* __restrict__ gh,
                              const float* __restrict__ b_ih, const float* __restrict__ b_hh,
                              float* __restrict__ h0, _Float16* __restrict__ h0f) {
    __shared__ float red[256];
    int r = blockIdx.x, j = threadIdx.x;
    size_t base = (size_t)r * 768;
    float ir = gi[base + j]            + b_ih[j];
    float iz = gi[base + HDIM + j]     + b_ih[HDIM + j];
    float ig = gi[base + 2 * HDIM + j] + b_ih[2 * HDIM + j];
    float hr = gh[base + j]            + b_hh[j];
    float hz = gh[base + HDIM + j]     + b_hh[HDIM + j];
    float hg = gh[base + 2 * HDIM + j] + b_hh[2 * HDIM + j];
    float rr = sigf(ir + hr);
    float zz = sigf(iz + hz);
    float gg = tanhf(ig + rr * hg);
    float hold = h0[(size_t)r * HDIM + j];
    float hn = (1.0f - zz) * gg + zz * hold;
    red[j] = hn * hn;
    __syncthreads();
    for (int s = 128; s > 0; s >>= 1) { if (j < s) red[j] += red[j + s]; __syncthreads(); }
    float inv = 1.0f / fmaxf(sqrtf(red[0]), 1e-12f);
    hn *= inv;
    h0[(size_t)r * HDIM + j]  = hn;
    h0f[(size_t)r * HDIM + j] = (_Float16)hn;
}

// Grid-stride edge pass with prefetch: pre[dst] += h[src] + h0[etype]; deg[dst]++
__global__ void edge_accum_kernel(const float* __restrict__ h, const float* __restrict__ h0,
                                  const int* __restrict__ src, const int* __restrict__ dst,
                                  const int* __restrict__ et, float* __restrict__ pre,
                                  float* __restrict__ deg) {
    int j = threadIdx.x;
    for (int e = blockIdx.x; e < NEDGE; e += gridDim.x) {
        int en = e + gridDim.x;
        if (en < NEDGE)
            __builtin_prefetch(&h[(size_t)src[en] * HDIM + j], 0, 3);
        int s = src[e], d = dst[e], r = et[e];
        float v = h[(size_t)s * HDIM + j] + h0[(size_t)r * HDIM + j];
        atomic_addf(&pre[(size_t)d * HDIM + j], v);
        if (j == 0) atomic_addf(&deg[d], 1.0f);
    }
}

// pre/deg -> f16 (mean folded before the GEMM; matmul distributes over it)
__global__ void scale_conv_kernel(const float* __restrict__ pre, const float* __restrict__ deg,
                                  _Float16* __restrict__ pref) {
    int gid = blockIdx.x * 256 + threadIdx.x;
    int n = gid >> 8;
    float d = fmaxf(deg[n], 1.0f);
    pref[gid] = (_Float16)(pre[gid] / d);
}

// Final pointwise: cur = G1 + G23[:, :256] -> rrelu -> l2norm;
// tw = sigmoid(G23[:, 256:] + b); h_new = tw*cur + (1-tw)*h
__global__ void ent_pw_kernel(const float* __restrict__ G1, const float* __restrict__ G23,
                              const float* __restrict__ tgb,
                              float* __restrict__ h, _Float16* __restrict__ hf,
                              float* __restrict__ out) {
    __shared__ float red[256];
    int n = blockIdx.x, j = threadIdx.x;
    size_t idx  = (size_t)n * HDIM + j;
    size_t idx2 = (size_t)n * 512 + j;
    float cur = G1[idx] + G23[idx2];
    cur = (cur >= 0.0f) ? cur : RRELU_SLOPE * cur;
    red[j] = cur * cur;
    __syncthreads();
    for (int s = 128; s > 0; s >>= 1) { if (j < s) red[j] += red[j + s]; __syncthreads(); }
    float inv = 1.0f / fmaxf(sqrtf(red[0]), 1e-12f);
    cur *= inv;
    float tw   = sigf(G23[idx2 + HDIM] + tgb[j]);
    float hold = h[idx];
    float hn   = tw * cur + (1.0f - tw) * hold;
    h[idx]   = hn;
    hf[idx]  = (_Float16)hn;
    out[idx] = hn;
}

// ---------------------------------------------------------------------------
extern "C" void kernel_launch(void* const* d_in, const int* in_sizes, int n_in,
                              void* d_out, int out_size, void* d_ws, size_t ws_size,
                              hipStream_t stream) {
    (void)in_sizes; (void)n_in; (void)out_size; (void)ws_size;
    const float* ent_emb  = (const float*)d_in[0];
    const float* emb_rel  = (const float*)d_in[1];
    const float* W_ih     = (const float*)d_in[2];   // [768, 512]
    const float* W_hh     = (const float*)d_in[3];   // [768, 256]
    const float* b_ih     = (const float*)d_in[4];
    const float* b_hh     = (const float*)d_in[5];
    const float* W_nb     = (const float*)d_in[6];   // [256, 256]
    const float* W_loop   = (const float*)d_in[7];
    const float* W_tg     = (const float*)d_in[8];
    const float* tg_bias  = (const float*)d_in[9];
    const int*   r_to_e   = (const int*)d_in[10];    // [T, M]
    const int*   r_to_e_r = (const int*)d_in[11];    // [T, M]
    const int*   src      = (const int*)d_in[12];    // [T, E]
    const int*   dst      = (const int*)d_in[13];
    const int*   etype    = (const int*)d_in[14];

    // ---- workspace carve-out (256B aligned) ----
    char* wp = (char*)d_ws;
    auto carve = [&](size_t bytes) -> void* {
        void* p = (void*)wp;
        wp += (bytes + 255) & ~(size_t)255;
        return p;
    };
    float*     h      = (float*)    carve((size_t)N_ENT * HDIM * 4);
    _Float16*  hf     = (_Float16*) carve((size_t)N_ENT * HDIM * 2);
    float*     h0     = (float*)    carve((size_t)NREL * HDIM * 4);
    _Float16*  h0f    = (_Float16*) carve((size_t)NRELP * HDIM * 2);
    _Float16*  xinf   = (_Float16*) carve((size_t)NRELP * 2 * HDIM * 2);
    float*     sums   = (float*)    carve(((size_t)NREL * HDIM + NRELP) * 4);
    float*     cnts   = sums + (size_t)NREL * HDIM;
    float*     gi     = (float*)    carve((size_t)NREL * 768 * 4);
    float*     gh     = (float*)    carve((size_t)NREL * 768 * 4);
    float*     pre    = (float*)    carve(((size_t)N_ENT * HDIM + N_ENT) * 4);
    float*     deg    = pre + (size_t)N_ENT * HDIM;
    _Float16*  pref   = (_Float16*) carve((size_t)N_ENT * HDIM * 2);
    float*     G23    = (float*)    carve((size_t)N_ENT * 512 * 4);
    _Float16*  Bp_ih  = (_Float16*) carve((size_t)512 * 768 * 2);
    _Float16*  Bp_hh  = (_Float16*) carve((size_t)256 * 768 * 2);
    _Float16*  Bp_nb  = (_Float16*) carve((size_t)256 * 256 * 2);
    _Float16*  Bp_cat = (_Float16*) carve((size_t)256 * 512 * 2);  // [W_loop | W_tg]

    dim3 wblk(32, 4);

    // ---- per-call init ----
    hipMemsetAsync(xinf, 0, (size_t)NRELP * 2 * HDIM * 2, stream);  // zero pad rows
    hipMemsetAsync(h0f,  0, (size_t)NRELP * HDIM * 2, stream);
    init_ent_kernel<<<N_ENT, 256, 0, stream>>>(ent_emb, h, hf);
    init_rel_kernel<<<NREL, 256, 0, stream>>>(emb_rel, h0, h0f);
    pack_b_kernel<<<(512 * 768 + 255) / 256, 256, 0, stream>>>(W_ih,   Bp_ih,  512, 768, 0,   768, 1);
    pack_b_kernel<<<(256 * 768 + 255) / 256, 256, 0, stream>>>(W_hh,   Bp_hh,  256, 768, 0,   768, 1);
    pack_b_kernel<<<(256 * 256 + 255) / 256, 256, 0, stream>>>(W_nb,   Bp_nb,  256, 256, 0,   256, 0);
    pack_b_kernel<<<(256 * 256 + 255) / 256, 256, 0, stream>>>(W_loop, Bp_cat, 256, 512, 0,   256, 0);
    pack_b_kernel<<<(256 * 256 + 255) / 256, 256, 0, stream>>>(W_tg,   Bp_cat, 256, 512, 256, 256, 0);

    for (int t = 0; t < TSEQ; ++t) {
        // ---- relation evolution ----
        hipMemsetAsync(sums, 0, ((size_t)NREL * HDIM + NRELP) * 4, stream);
        seg_accum_kernel<<<2048, 256, 0, stream>>>(h, r_to_e + (size_t)t * NRTE,
                                                   r_to_e_r + (size_t)t * NRTE, sums, cnts);
        build_xin_kernel<<<NREL, 256, 0, stream>>>(emb_rel, sums, cnts, xinf);
        wmma_gemm4_kernel<768, 512, 512, 768><<<dim3(3, 32), wblk, 0, stream>>>(xinf, Bp_ih, gi, NREL);
        wmma_gemm4_kernel<768, 256, 256, 768><<<dim3(3, 32), wblk, 0, stream>>>(h0f,  Bp_hh, gh, NREL);
        gru_pw_kernel<<<NREL, 256, 0, stream>>>(gi, gh, b_ih, b_hh, h0, h0f);

        // ---- entity evolution (matmul pulled out of the edge loop) ----
        hipMemsetAsync(pre, 0, ((size_t)N_ENT * HDIM + N_ENT) * 4, stream);
        edge_accum_kernel<<<2048, 256, 0, stream>>>(h, h0, src + (size_t)t * NEDGE,
                                                    dst + (size_t)t * NEDGE,
                                                    etype + (size_t)t * NEDGE, pre, deg);
        scale_conv_kernel<<<(N_ENT * HDIM) / 256, 256, 0, stream>>>(pre, deg, pref);

        float* out_t = (float*)d_out + (size_t)t * N_ENT * HDIM;
        // G1 = (pre/deg) @ W_nb  (reuses pre f32 buffer; A already in f16 copy)
        wmma_gemm4_kernel<256, 256, 256, 256><<<dim3(1, 1250), wblk, 0, stream>>>(pref, Bp_nb, pre, N_ENT);
        // G23 = h @ [W_loop | W_tg]  (single pass over h_f16)
        wmma_gemm4_kernel<512, 256, 256, 512><<<dim3(2, 1250), wblk, 0, stream>>>(hf, Bp_cat, G23, N_ENT);

        ent_pw_kernel<<<N_ENT, 256, 0, stream>>>(pre, G23, tg_bias, h, hf, out_t);
    }
}
